// GeneralizedMHDPAttentionModule_11682311045239
// MI455X (gfx1250) — matrix operand order repaired and checked by reading, not compiled, not run on hardware
//
#include <hip/hip_runtime.h>
#include <hip/hip_bf16.h>
#include <math.h>

typedef __attribute__((ext_vector_type(16))) __bf16 v16bf;
typedef __attribute__((ext_vector_type(8)))  float  v8f;

#define B_   4
#define N_   1024
#define M_   4096
#define H_   8
#define D_   64
#define DF_  512
#define EPS_ 1e-8f
#define GRP_ 4      // (b,h) slices per L2-resident group: 4 * 16.8MB = 67MB << 192MB L2
#define NCH_ 8      // column-stat row chunks (1024/128)

__device__ __forceinline__ v8f wmma_bf16(v16bf a, v16bf b, v8f c) {
  // D = A(16x32 bf16) * B(32x16 bf16) + C(16x16 f32)
  return __builtin_amdgcn_wmma_f32_16x16x32_bf16(false, a, false, b, (short)0, c,
                                                 false, false);
}

// ---------------------------------------------------------------------------
// Generic projection GEMM: C[r,c] = sum_k A[r,k] * W[k,c] (K = 512).
// One wave computes a 16x64 output tile (4 accumulators); f32 inputs are
// converted to bf16 while building fragments; output bf16 or f32.
// ---------------------------------------------------------------------------
template <bool BF16OUT>
__global__ void proj_gemm(const float* __restrict__ A, const float* __restrict__ W,
                          __bf16* __restrict__ Cb, float* __restrict__ Cf) {
  const int wid   = blockIdx.x * 4 + (threadIdx.x >> 5);
  const int lane  = threadIdx.x & 31;
  const int tr    = wid >> 3;
  const int tc4   = wid & 7;
  const int lrow  = lane & 15;
  const int khalf = lane >> 4;
  const int c0    = tc4 * 64;

  const float* Arow = A + (size_t)(tr * 16 + lrow) * DF_;
  v8f acc[4] = {};

  for (int k0 = 0; k0 < DF_; k0 += 32) {
    v16bf a;
    const float* ap = Arow + k0 + 16 * khalf;
#pragma unroll
    for (int v = 0; v < 8; ++v) {
      const int kk = 2 * v;
      a[2 * v]     = (__bf16)ap[kk];
      a[2 * v + 1] = (__bf16)ap[kk + 1];
    }
#pragma unroll
    for (int t = 0; t < 4; ++t) {
      const int col   = c0 + t * 16 + lrow;
      const float* wp = W + (size_t)(k0 + 16 * khalf) * DF_ + col;
      v16bf bfrag;
#pragma unroll
      for (int e = 0; e < 16; ++e) bfrag[e] = (__bf16)wp[(size_t)e * DF_];
      acc[t] = wmma_bf16(a, bfrag, acc[t]);
    }
  }

  const int rbase = 8 * khalf;
#pragma unroll
  for (int t = 0; t < 4; ++t)
#pragma unroll
    for (int rr = 0; rr < 8; ++rr) {
      const size_t idx = (size_t)(tr * 16 + rbase + rr) * DF_ + c0 + t * 16 + lrow;
      if (BF16OUT) Cb[idx] = (__bf16)acc[t][rr];
      else         Cf[idx] = acc[t][rr];
    }
}

// ---------------------------------------------------------------------------
// Scores for one slice group: S[bh][n,m] = sf * sum_d Q.K. Wave = 16x64 tile.
// Default (RT) stores: scores must stay L2-resident for the next 4 kernels.
// ---------------------------------------------------------------------------
__global__ void scores_kernel(const __bf16* __restrict__ Qb,
                              const __bf16* __restrict__ Kb,
                              float* __restrict__ S, int bh0) {
  const int wid   = blockIdx.x * 4 + (threadIdx.x >> 5);
  const int lane  = threadIdx.x & 31;
  const int tm4   = wid & 63;
  const int tn    = (wid >> 6) & 63;
  const int bh    = bh0 + blockIdx.y;
  const int b     = bh >> 3, h = bh & 7;
  const int n0    = tn * 16, m0 = tm4 * 64;
  const int lrow  = lane & 15, khalf = lane >> 4;

  const __bf16* Qrow = Qb + ((size_t)(b * N_) + n0 + lrow) * DF_ + h * D_;
  v8f acc[4] = {};

  for (int d0 = 0; d0 < D_; d0 += 32) {
    v16bf a;
    const __bf16* qa = Qrow + d0 + 16 * khalf;
#pragma unroll
    for (int v = 0; v < 8; ++v) {
      const int kk = 2 * v;
      a[2 * v]     = qa[kk];
      a[2 * v + 1] = qa[kk + 1];
    }
#pragma unroll
    for (int t = 0; t < 4; ++t) {
      const int mcol   = m0 + t * 16 + lrow;
      const __bf16* kr = Kb + ((size_t)(b * M_) + mcol) * DF_ + h * D_ + d0 + 16 * khalf;
      v16bf bfrag;
#pragma unroll
      for (int e = 0; e < 16; ++e) bfrag[e] = kr[e];
      acc[t] = wmma_bf16(a, bfrag, acc[t]);
    }
  }

  const float sf = 0.125f;  // 1/sqrt(64)
  float* Sbh = S + (size_t)bh * N_ * M_;
  const int rbase = 8 * khalf;
#pragma unroll
  for (int t = 0; t < 4; ++t)
#pragma unroll
    for (int rr = 0; rr < 8; ++rr)
      Sbh[(size_t)(n0 + rbase + rr) * M_ + m0 + t * 16 + lrow] = acc[t][rr] * sf;
}

// ---------------------------------------------------------------------------
// Partial online-softmax stats over the QUERY axis n, 128-row chunks so the
// per-group grid stays wide (16 x 8 x GRP blocks of 256 threads).
// ---------------------------------------------------------------------------
__global__ void colpart_kernel(const float* __restrict__ S,
                               float* __restrict__ pmax, float* __restrict__ psum,
                               int bh0) {
  const int m  = blockIdx.x * 256 + threadIdx.x;
  const int nc = blockIdx.y;   // row chunk
  const int sl = blockIdx.z;   // slice within group
  const float* p = S + (size_t)(bh0 + sl) * N_ * M_ + (size_t)(nc * 128) * M_ + m;
  float mx = -3.4e38f, sm = 0.f;
  for (int n = 0; n < 128; ++n) {
    const float x = p[(size_t)n * M_];
    if (x > mx) { sm *= __expf(mx - x); mx = x; }
    sm += __expf(x - mx);
  }
  const size_t o = ((size_t)sl * NCH_ + nc) * M_ + m;
  pmax[o] = mx;
  psum[o] = sm;
}

// Merge the NCH_ partials per column into colmax/colsum.
__global__ void colcombine_kernel(const float* __restrict__ pmax,
                                  const float* __restrict__ psum,
                                  float* __restrict__ colmax,
                                  float* __restrict__ colsum, int bh0) {
  const int t  = blockIdx.x * blockDim.x + threadIdx.x;  // GRP_*M_ threads
  const int sl = t >> 12;
  const int m  = t & (M_ - 1);
  float mx = -3.4e38f;
#pragma unroll
  for (int c = 0; c < NCH_; ++c)
    mx = fmaxf(mx, pmax[((size_t)sl * NCH_ + c) * M_ + m]);
  float sm = 0.f;
#pragma unroll
  for (int c = 0; c < NCH_; ++c) {
    const size_t o = ((size_t)sl * NCH_ + c) * M_ + m;
    sm += psum[o] * __expf(pmax[o] - mx);
  }
  colmax[(size_t)(bh0 + sl) * M_ + m] = mx;
  colsum[(size_t)(bh0 + sl) * M_ + m] = sm;
}

// ---------------------------------------------------------------------------
// attn0[n,m] = exp(S - colmax[m]) / colsum[m] (in place, L2-resident), plus
// rowrecip[n] = 1/(sum_m attn0 + eps). One block per (slice, n) row.
// ---------------------------------------------------------------------------
__global__ void expnorm_kernel(float* __restrict__ S,
                               const float* __restrict__ colmax,
                               const float* __restrict__ colsum,
                               float* __restrict__ rowrecip, int bh0) {
  const int bh = bh0 + (blockIdx.x >> 10);
  const int n  = blockIdx.x & (N_ - 1);
  float* row      = S + ((size_t)bh * N_ + n) * M_;
  const float* cm = colmax + (size_t)bh * M_;
  const float* cs = colsum + (size_t)bh * M_;
  float local = 0.f;
  for (int m = threadIdx.x; m < M_; m += blockDim.x) {
    const float e = __expf(row[m] - cm[m]) / cs[m];
    row[m] = e;
    local += e;
  }
  __shared__ float red[256];
  red[threadIdx.x] = local;
  __syncthreads();
  for (int s = 128; s > 0; s >>= 1) {
    if (threadIdx.x < s) red[threadIdx.x] += red[threadIdx.x + s];
    __syncthreads();
  }
  if (threadIdx.x == 0)
    rowrecip[(size_t)bh * N_ + n] = 1.0f / (red[0] + EPS_);
}

// ---------------------------------------------------------------------------
// Fused AV GEMM + attn finalization, split over 4 key ranges for occupancy:
//   outheads[b,n,h,:] += rowrecip[n] * sum_{m in range} attn0[n,m] * V[b,m,h,:]
//   attn[n,m]          = attn0[n,m] * rowrecip[n]
// attn0 is read once (NT load) and the final attn written in place (NT store)
// so the 536MB streaming output does not evict the L2-resident group.
// Partial outheads accumulated with global f32 atomic adds (row scaling is
// linear, so per-range partials scale independently).
// ---------------------------------------------------------------------------
__global__ void av_kernel(float* __restrict__ S, const __bf16* __restrict__ Vb,
                          const float* __restrict__ rowrecip,
                          float* __restrict__ outheads, int bh0) {
  __shared__ __bf16 Vt[32][D_ + 2];
  const int bh   = bh0 + blockIdx.y;
  const int b    = bh >> 3, h = bh & 7;
  const int wid  = threadIdx.x >> 5;
  const int lane = threadIdx.x & 31;
  const int n0   = (blockIdx.x * 4 + wid) * 16;
  const int mlo  = blockIdx.z * (M_ / 4);
  const int mhi  = mlo + (M_ / 4);
  const int lrow = lane & 15, khalf = lane >> 4;
  const int arow = n0 + lrow;

  float* Srow         = S + ((size_t)bh * N_ + arow) * M_;
  const float rcp_row = rowrecip[(size_t)bh * N_ + arow];
  v8f acc[4] = {};

  for (int m0 = mlo; m0 < mhi; m0 += 32) {
    __syncthreads();
    for (int i = threadIdx.x; i < 32 * D_; i += blockDim.x) {
      const int r = i >> 6, c = i & (D_ - 1);
      Vt[r][c] = Vb[((size_t)b * M_ + m0 + r) * DF_ + h * D_ + c];
    }
    __syncthreads();

    v16bf a;
    const int mb = m0 + 16 * khalf;
#pragma unroll
    for (int v = 0; v < 8; ++v) {
      const int kk   = 2 * v;
      const float x0 = __builtin_nontemporal_load(&Srow[mb + kk]);
      const float x1 = __builtin_nontemporal_load(&Srow[mb + kk + 1]);
      a[2 * v]     = (__bf16)x0;
      a[2 * v + 1] = (__bf16)x1;
      __builtin_nontemporal_store(x0 * rcp_row, &Srow[mb + kk]);
      __builtin_nontemporal_store(x1 * rcp_row, &Srow[mb + kk + 1]);
    }
#pragma unroll
    for (int t = 0; t < 4; ++t) {
      v16bf bfrag;
#pragma unroll
      for (int e = 0; e < 16; ++e) bfrag[e] = Vt[16 * khalf + e][t * 16 + lrow];
      acc[t] = wmma_bf16(a, bfrag, acc[t]);
    }
  }

  const int rbase = 8 * khalf;
#pragma unroll
  for (int rr = 0; rr < 8; ++rr) {
    const int r     = n0 + rbase + rr;
    const float rcp = rowrecip[(size_t)bh * N_ + r];
#pragma unroll
    for (int t = 0; t < 4; ++t)
      atomicAdd(&outheads[((size_t)b * N_ + r) * DF_ + h * D_ + t * 16 + lrow],
                acc[t][rr] * rcp);
  }
}

__global__ void zero_kernel(float4* __restrict__ p) {
  p[blockIdx.x * blockDim.x + threadIdx.x] = make_float4(0.f, 0.f, 0.f, 0.f);
}

// ---------------------------------------------------------------------------
extern "C" void kernel_launch(void* const* d_in, const int* in_sizes, int n_in,
                              void* d_out, int out_size, void* d_ws, size_t ws_size,
                              hipStream_t stream) {
  (void)in_sizes; (void)n_in; (void)out_size; (void)ws_size;
  const float* f1 = (const float*)d_in[0];   // [4,1024,512]
  const float* f2 = (const float*)d_in[1];   // [4,4096,512]
  const float* Wq = (const float*)d_in[2];   // [512,512]
  const float* Wk = (const float*)d_in[3];
  const float* Wv = (const float*)d_in[4];
  const float* Wo = (const float*)d_in[5];

  float* updates = (float*)d_out;                    // [4,1024,512]
  float* attn    = updates + (size_t)B_ * N_ * DF_;  // [4,8,1024,4096] (scores scratch)

  char* ws = (char*)d_ws;
  __bf16* Qb      = (__bf16*)(ws);               //  4 MB  [B*N, 512] bf16
  __bf16* Kb      = (__bf16*)(ws + 4194304);     // 16 MB  [B*M, 512] bf16
  __bf16* Vb      = (__bf16*)(ws + 20971520);    // 16 MB  [B*M, 512] bf16
  float* colmax   = (float*)(ws + 37748736);     // 512 KB [B*H, M]
  float* colsum   = (float*)(ws + 38273024);     // 512 KB [B*H, M]
  float* rowrecip = (float*)(ws + 38797312);     // 128 KB [B*H, N]
  float* outheads = (float*)(ws + 38928384);     //  8 MB  [B*N, 512] f32
  float* pmax     = (float*)(ws + 47316992);     // 512 KB [GRP, NCH, M]
  float* psum     = (float*)(ws + 47841280);     // 512 KB [GRP, NCH, M]

  // outheads accumulated via atomics -> zero it (d_ws is poisoned, not zeroed)
  zero_kernel<<<(B_ * N_ * DF_ / 4) / 256, 256, 0, stream>>>((float4*)outheads);

  // Q/K/V projections (f32 in, bf16 out)
  proj_gemm<true><<<(B_ * N_ / 16) * 8 / 4, 128, 0, stream>>>(f1, Wq, Qb, nullptr);
  proj_gemm<true><<<(B_ * M_ / 16) * 8 / 4, 128, 0, stream>>>(f2, Wk, Kb, nullptr);
  proj_gemm<true><<<(B_ * M_ / 16) * 8 / 4, 128, 0, stream>>>(f2, Wv, Vb, nullptr);

  // L2-blocked softmax/AV pipeline: 4 (b,h) slices (67MB of scores) at a time
  // stay resident in the 192MB L2 across the 4 dependent kernels; only the
  // final NT attn writes stream to HBM.
  for (int bh0 = 0; bh0 < B_ * H_; bh0 += GRP_) {
    dim3 sg((N_ / 16) * (M_ / 64) / 4, GRP_);
    scores_kernel<<<sg, 128, 0, stream>>>(Qb, Kb, attn, bh0);
    dim3 cp(M_ / 256, NCH_, GRP_);
    colpart_kernel<<<cp, 256, 0, stream>>>(attn, pmax, psum, bh0);
    colcombine_kernel<<<GRP_ * M_ / 256, 256, 0, stream>>>(pmax, psum, colmax, colsum, bh0);
    expnorm_kernel<<<GRP_ * N_, 256, 0, stream>>>(attn, colmax, colsum, rowrecip, bh0);
    dim3 avg(N_ / 16 / 4, GRP_, 4);
    av_kernel<<<avg, 128, 0, stream>>>(attn, Vb, rowrecip, outheads, bh0);
  }

  // updates = outheads @ Wo (f32 out)
  proj_gemm<false><<<(B_ * N_ / 16) * 8 / 4, 128, 0, stream>>>(outheads, Wo, nullptr, updates);
}
